// TrellisExpertMLP_45887430590500
// MI455X (gfx1250) — matrix operand order repaired and checked by reading, not compiled, not run on hardware
//
#include <hip/hip_runtime.h>
#include <hip/hip_bf16.h>

typedef __attribute__((ext_vector_type(16))) __bf16 v16bf;
typedef __attribute__((ext_vector_type(8)))  float  v8f;
typedef __attribute__((ext_vector_type(4)))  unsigned int v4u;
typedef __attribute__((ext_vector_type(8)))  int v8i;
typedef __attribute__((ext_vector_type(4)))  int v4i;

#define HIDDEN 4096
#define INTER  14336
#define TOKENS 4096

// Tiling: block = 128(M) x 64(N), K-step 32, 8 waves (4 M x 2 N), wave tile 32x32.
#define BM 128
#define BN 64
#define BK 32
#define XS_STRIDE 40   // bf16 elems per LDS row of 128x32 A tile (80B: 16B-aligned, bank-skewed)
#define WS_STRIDE 72   // bf16 elems per LDS row of 32x64 B tile (144B: 16B-aligned, bank-skewed)

#if __has_builtin(__builtin_amdgcn_tensor_load_to_lds) && __has_builtin(__builtin_amdgcn_s_wait_tensorcnt)
#define USE_TDM 1
#else
#define USE_TDM 0
#endif

__device__ __forceinline__ void cvt8_store(__bf16* dst, float4 f0, float4 f1) {
  __bf16 o[8] __attribute__((aligned(16)));
  o[0] = (__bf16)f0.x; o[1] = (__bf16)f0.y; o[2] = (__bf16)f0.z; o[3] = (__bf16)f0.w;
  o[4] = (__bf16)f1.x; o[5] = (__bf16)f1.y; o[6] = (__bf16)f1.z; o[7] = (__bf16)f1.w;
  *(uint4*)dst = *(const uint4*)o;
}

__device__ __forceinline__ float fast_rcp(float v) {
#if __has_builtin(__builtin_amdgcn_rcpf)
  return __builtin_amdgcn_rcpf(v);   // v_rcp_f32
#else
  return 1.0f / v;
#endif
}

#if USE_TDM
// Tensor Data Mover: DMA a 128-row x 32-col bf16 tile (row stride = INTER elems)
// from global into LDS, hardware-padding each 64B row by 16B to land on the
// XS_STRIDE=40-elem (80B) bank-skewed LDS pitch. D# packing per ISA 08 §8.3/8.4.
__device__ __forceinline__ void tdm_load_h_tile(const __bf16* gsrc, unsigned lds_off) {
  const unsigned long long ga = (unsigned long long)(size_t)gsrc;
  v4u g0;
  g0.x = 1u;                                   // count=1, user descriptor
  g0.y = lds_off;                              // lds_addr (bytes)
  g0.z = (unsigned)(ga & 0xffffffffu);         // global_addr[31:0]
  g0.w = (unsigned)((ga >> 32) & 0x1ffffffu)   // global_addr[56:32]
       | (2u << 30);                           // type = 2 ("image")
  v8i g1 = {};
  g1[0] = (int)((1u << 16)                     // data_size = 2 bytes (bf16)
              | (1u << 20)                     // pad_enable (LDS dest padding)
              | (3u << 22)                     // pad_interval: 16 DWORDs (one 64B row)
              | (3u << 25));                   // pad_amount:   4 DWORDs (16B)
  g1[1] = (int)(((unsigned)INTER & 0xffffu) << 16);       // tensor_dim0[15:0]  @ bits 63:48
  g1[2] = (int)((((unsigned)INTER >> 16) & 0xffffu)       // tensor_dim0[31:16]
              | (((unsigned)TOKENS & 0xffffu) << 16));    // tensor_dim1[15:0]
  g1[3] = (int)((((unsigned)TOKENS >> 16) & 0xffffu)      // tensor_dim1[31:16]
              | (32u << 16));                             // tile_dim0 = 32
  g1[4] = (int)128u;                                      // tile_dim1 = 128
  g1[5] = (int)INTER;                                     // tensor_dim0_stride[31:0]
  // g1[6], g1[7]: stride high bits / dim1 stride = 0 (2D tile)
  v4i gz = {};
#if defined(__clang_major__) && (__clang_major__ >= 23)
  v8i gz8 = {};
  __builtin_amdgcn_tensor_load_to_lds(g0, g1, gz, gz, gz8, 0);
#else
  __builtin_amdgcn_tensor_load_to_lds(g0, g1, gz, gz, 0);
#endif
}
#endif

// ---------------------------------------------------------------------------
// Kernel 1: fused h = silu(x @ Wg) * (x @ Wu), h stored as bf16 [TOKENS, INTER]
// x is converted fp32->bf16 during LDS staging (no pre-pass).
// blockIdx.x = M tile (fastest varying) so concurrent blocks share the same
// N-stripe of weights -> weights cross HBM once, x re-read from L2.
// ---------------------------------------------------------------------------
__global__ __launch_bounds__(256)
void swiglu_gateup(const float* __restrict__ x,
                   const float* __restrict__ wg,
                   const float* __restrict__ wu,
                   __bf16* __restrict__ h) {
  __shared__ __bf16 xs[2][BM * XS_STRIDE];       // 2 x 10.0 KB
  __shared__ __bf16 wls[2][2][BK * WS_STRIDE];   // 2 x 9.0 KB (gate, up)

  const int tid   = threadIdx.x;
  const int lane  = tid & 31;
  const int wave  = tid >> 5;
  const int wm    = wave & 3;
  const int wn    = wave >> 2;
  const int m0    = blockIdx.x * BM;
  const int n0    = blockIdx.y * BN;
  const int lrow  = lane & 15;
  const int lhalf = lane >> 4;

  const int xrow = tid >> 1, xhalf = tid & 1;
  const float* xsrc = x + (size_t)(m0 + xrow) * HIDDEN + xhalf * 16;
  const int wmat = tid >> 7;
  const int widx = tid & 127;
  const int wrow = widx >> 2, wseg = widx & 3;
  const float* wsrc = (wmat ? wu : wg) + (size_t)wrow * INTER + n0 + wseg * 16;

  v8f accg[2][2] = {};
  v8f accu[2][2] = {};

  float4 xr0, xr1, xr2, xr3;
  float4 wr0, wr1, wr2, wr3;

  auto fetch = [&](int k0) {
    const float* xp = xsrc + k0;
    xr0 = ((const float4*)xp)[0]; xr1 = ((const float4*)xp)[1];
    xr2 = ((const float4*)xp)[2]; xr3 = ((const float4*)xp)[3];
    const float* wp = wsrc + (size_t)k0 * INTER;
    wr0 = ((const float4*)wp)[0]; wr1 = ((const float4*)wp)[1];
    wr2 = ((const float4*)wp)[2]; wr3 = ((const float4*)wp)[3];
  };

  fetch(0);
  int buf = 0;
  for (int k0 = 0; k0 < HIDDEN; k0 += BK) {
    {
      __bf16* xd = &xs[buf][xrow * XS_STRIDE + xhalf * 16];
      cvt8_store(xd,     xr0, xr1);
      cvt8_store(xd + 8, xr2, xr3);
      __bf16* wd = &wls[buf][wmat][wrow * WS_STRIDE + wseg * 16];
      cvt8_store(wd,     wr0, wr1);
      cvt8_store(wd + 8, wr2, wr3);
    }
    __syncthreads();

    if (k0 + BK < HIDDEN) fetch(k0 + BK);
    if (k0 + 4 * BK < HIDDEN)
      __builtin_prefetch(wsrc + (size_t)(k0 + 4 * BK) * INTER, 0, 1);

    // A fragments per ISA 16-bit 16x32 layout.
    v16bf a[2];
    #pragma unroll
    for (int mi = 0; mi < 2; ++mi) {
      const __bf16* base = &xs[buf][(wm * 32 + mi * 16 + lrow) * XS_STRIDE];
      #pragma unroll
      for (int p = 0; p < 8; ++p) {
        const int kk = ((p & 3) << 1) + ((p >> 2) << 4) + (lhalf << 3);
        a[mi][2 * p]     = base[kk];
        a[mi][2 * p + 1] = base[kk + 1];
      }
    }
    // B fragments: lane = N column, 16 K per lane in table order.
    v16bf bg[2], bu[2];
    #pragma unroll
    for (int ni = 0; ni < 2; ++ni) {
      const __bf16* gp = &wls[buf][0][(lhalf << 3) * WS_STRIDE + wn * 32 + ni * 16 + lrow];
      const __bf16* up = &wls[buf][1][(lhalf << 3) * WS_STRIDE + wn * 32 + ni * 16 + lrow];
      #pragma unroll
      for (int e = 0; e < 16; ++e) { const int kr = (e & 7) + ((e >> 3) << 4); bg[ni][e] = gp[kr * WS_STRIDE]; bu[ni][e] = up[kr * WS_STRIDE]; }
    }

    #pragma unroll
    for (int mi = 0; mi < 2; ++mi)
      #pragma unroll
      for (int ni = 0; ni < 2; ++ni) {
        accg[mi][ni] = __builtin_amdgcn_wmma_f32_16x16x32_bf16(
            false, a[mi], false, bg[ni], (short)0, accg[mi][ni], false, false);
        accu[mi][ni] = __builtin_amdgcn_wmma_f32_16x16x32_bf16(
            false, a[mi], false, bu[ni], (short)0, accu[mi][ni], false, false);
      }
    buf ^= 1;
  }

  #pragma unroll
  for (int mi = 0; mi < 2; ++mi)
    #pragma unroll
    for (int ni = 0; ni < 2; ++ni)
      #pragma unroll
      for (int i = 0; i < 8; ++i) {
        const int row = m0 + wm * 32 + mi * 16 + i + (lhalf << 3);
        const int col = n0 + wn * 32 + ni * 16 + lrow;
        const float g = accg[mi][ni][i];
        const float u = accu[mi][ni][i];
        const float s = g * fast_rcp(1.0f + __expf(-g)) * u;
        h[(size_t)row * INTER + col] = (__bf16)s;
      }
}

// ---------------------------------------------------------------------------
// Kernel 2: out = h @ Wdown. h tile (pure bf16) is DMA'd into LDS by the
// Tensor Data Mover (one tensor_load_to_lds per K-step, issued by wave 0,
// drained with s_wait_tensorcnt); weights keep the register path (need cvt).
// ---------------------------------------------------------------------------
__global__ __launch_bounds__(256)
void swiglu_down(const __bf16* __restrict__ h,
                 const float*  __restrict__ wd,
                 float* __restrict__ out) {
  __shared__ __bf16 hs[2][BM * XS_STRIDE];
  __shared__ __bf16 wls[2][BK * WS_STRIDE];

  const int tid   = threadIdx.x;
  const int lane  = tid & 31;
  const int wave  = tid >> 5;
  const int wm    = wave & 3;
  const int wn    = wave >> 2;
  const int m0    = blockIdx.x * BM;
  const int n0    = blockIdx.y * BN;
  const int lrow  = lane & 15;
  const int lhalf = lane >> 4;

  const int wrow = tid >> 3, wseg = tid & 7;
  const float* wsrc = wd + (size_t)wrow * HIDDEN + n0 + wseg * 8;

  v8f acc[2][2] = {};
  float4 wr0, wr1;
  auto fetch_w = [&](int k0) {
    const float* wp = wsrc + (size_t)k0 * HIDDEN;
    wr0 = ((const float4*)wp)[0];
    wr1 = ((const float4*)wp)[1];
  };

#if USE_TDM
  const __bf16* htile = h + (size_t)m0 * INTER;
  if (wave == 0) tdm_load_h_tile(htile, (unsigned)(size_t)&hs[0][0]);
#else
  const int hrow = tid >> 1, hhalf = tid & 1;
  const __bf16* hsrc = h + (size_t)(m0 + hrow) * INTER + hhalf * 16;
  uint4 hr;
  auto fetch_h = [&](int k0) { hr = *(const uint4*)(hsrc + k0); };
  fetch_h(0);
#endif
  fetch_w(0);

  int buf = 0;
  for (int k0 = 0; k0 < INTER; k0 += BK) {
    cvt8_store(&wls[buf][wrow * WS_STRIDE + wseg * 8], wr0, wr1);
#if USE_TDM
    if (wave == 0) __builtin_amdgcn_s_wait_tensorcnt(0);  // hs[buf] DMA complete
#else
    *(uint4*)(&hs[buf][hrow * XS_STRIDE + hhalf * 16]) = hr;
#endif
    __syncthreads();

    if (k0 + BK < INTER) {
      fetch_w(k0 + BK);
#if USE_TDM
      if (wave == 0)
        tdm_load_h_tile(htile + (k0 + BK), (unsigned)(size_t)&hs[buf ^ 1][0]);
#else
      fetch_h(k0 + BK);
#endif
    }
    if (k0 + 4 * BK < INTER)
      __builtin_prefetch(wsrc + (size_t)(k0 + 4 * BK) * HIDDEN, 0, 1);

    v16bf a[2];
    #pragma unroll
    for (int mi = 0; mi < 2; ++mi) {
      const __bf16* base = &hs[buf][(wm * 32 + mi * 16 + lrow) * XS_STRIDE];
      #pragma unroll
      for (int p = 0; p < 8; ++p) {
        const int kk = ((p & 3) << 1) + ((p >> 2) << 4) + (lhalf << 3);
        a[mi][2 * p]     = base[kk];
        a[mi][2 * p + 1] = base[kk + 1];
      }
    }
    v16bf b[2];
    #pragma unroll
    for (int ni = 0; ni < 2; ++ni) {
      const __bf16* bp = &wls[buf][(lhalf << 3) * WS_STRIDE + wn * 32 + ni * 16 + lrow];
      #pragma unroll
      for (int e = 0; e < 16; ++e) b[ni][e] = bp[((e & 7) + ((e >> 3) << 4)) * WS_STRIDE];
    }

    #pragma unroll
    for (int mi = 0; mi < 2; ++mi)
      #pragma unroll
      for (int ni = 0; ni < 2; ++ni)
        acc[mi][ni] = __builtin_amdgcn_wmma_f32_16x16x32_bf16(
            false, a[mi], false, b[ni], (short)0, acc[mi][ni], false, false);
    buf ^= 1;
  }

  #pragma unroll
  for (int mi = 0; mi < 2; ++mi)
    #pragma unroll
    for (int ni = 0; ni < 2; ++ni)
      #pragma unroll
      for (int i = 0; i < 8; ++i) {
        const int row = m0 + wm * 32 + mi * 16 + i + (lhalf << 3);
        const int col = n0 + wn * 32 + ni * 16 + lrow;
        __builtin_nontemporal_store(acc[mi][ni][i], &out[(size_t)row * HIDDEN + col]);
      }
}

// ---------------------------------------------------------------------------
extern "C" void kernel_launch(void* const* d_in, const int* in_sizes, int n_in,
                              void* d_out, int out_size, void* d_ws, size_t ws_size,
                              hipStream_t stream) {
  (void)in_sizes; (void)n_in; (void)out_size; (void)ws_size;
  const float* x  = (const float*)d_in[0];
  const float* wg = (const float*)d_in[1];
  const float* wu = (const float*)d_in[2];
  const float* wd = (const float*)d_in[3];

  __bf16* h = (__bf16*)d_ws;   // 117.4 MB bf16 intermediate

  swiglu_gateup<<<dim3(TOKENS / BM, INTER / BN), 256, 0, stream>>>(x, wg, wu, h);
  swiglu_down  <<<dim3(TOKENS / BM, HIDDEN / BN), 256, 0, stream>>>(h, wd, (float*)d_out);
}
